// SpatialSpectralFeatureMarking_33827162423566
// MI455X (gfx1250) — compile-verified, matrix operand order checked by reading, not run on hardware
//
#include <hip/hip_runtime.h>
#include <stdint.h>

// ---------------------------------------------------------------------------
// SpatialSpectralFeatureMarking for MI455X (gfx1250, wave32, WMMA)
// B=8, C=64, OUT=32, K=3, H=W=64
// ---------------------------------------------------------------------------

typedef __attribute__((ext_vector_type(16))) _Float16 v16h;
typedef __attribute__((ext_vector_type(8)))  _Float16 v8h;
typedef __attribute__((ext_vector_type(8)))  float    v8f;
typedef __attribute__((ext_vector_type(4)))  int      v4i;

#define BSZ   8
#define CCH   64
#define OUTC  32
#define HH    64
#define WW    64
#define HW    4096           // 64*64
#define KPAD  160            // 144 (=16*9) padded to 5 * 32 WMMA K-steps
#define SROW  168            // LDS staging row stride (halves) = 84 dwords
#define FCH   128            // fused channels (2C)

static __device__ __forceinline__ float silu_f(float v)    { return v / (1.0f + __expf(-v)); }
static __device__ __forceinline__ float sigmoid_f(float v) { return 1.0f / (1.0f + __expf(-v)); }

// ---- optional CDNA5 async global->LDS path (compile-safe guard) -----------
#if defined(__gfx1250__) && \
    __has_builtin(__builtin_amdgcn_global_load_async_to_lds_b128) && \
    __has_builtin(__builtin_amdgcn_s_wait_asynccnt)
#define HAVE_ASYNC_LDS 1
#define GLOBAL_AS __attribute__((address_space(1)))
#define LDS_AS    __attribute__((address_space(3)))
#endif

// Stage `nhalves` f16 (multiple of 8) from global to LDS, block-cooperative.
static __device__ __forceinline__ void stage_to_lds(const _Float16* __restrict__ g,
                                                    _Float16* __restrict__ l,
                                                    int nhalves, int tid, int nthr) {
#if defined(HAVE_ASYNC_LDS)
    for (int i = tid; i < nhalves / 8; i += nthr) {
        __builtin_amdgcn_global_load_async_to_lds_b128(
            (GLOBAL_AS v4i*)(unsigned long long)(uintptr_t)(g + i * 8),
            (LDS_AS v4i*)(unsigned)(uintptr_t)(l + i * 8), 0, 0);
    }
    __builtin_amdgcn_s_wait_asynccnt(0);
#else
    for (int i = tid; i < nhalves / 8; i += nthr)
        *(uint4*)(l + i * 8) = *(const uint4*)(g + i * 8);
#endif
    __syncthreads();
}

// ---------------------------------------------------------------------------
// Kernel P: convert + pad GEMM weights to f16
// ---------------------------------------------------------------------------
__global__ void prep_weights_kernel(const float* __restrict__ oc_w,
                                    const float* __restrict__ fu_w,
                                    _Float16* __restrict__ wA,
                                    _Float16* __restrict__ wF) {
    int tid = threadIdx.x;
    for (int i = tid; i < 4 * 16 * KPAD; i += 256) {
        int g = i / (16 * KPAD);
        int r = i % (16 * KPAD);
        int o = r / KPAD;
        int k = r % KPAD;
        float v = (k < 144) ? oc_w[(g * 16 + o) * 144 + k] : 0.0f;
        wA[i] = (_Float16)v;
    }
    for (int i = tid; i < OUTC * FCH; i += 256)
        wF[i] = (_Float16)fu_w[i];
}

// ---------------------------------------------------------------------------
// Kernel A1: RF attention maps, restructured for coalesced output.
// Block = one (b, group, image row): 64 positions x 16 channels x 9 taps.
// Output rows [p][160] are contiguous in global, flushed via b128 stores.
// ---------------------------------------------------------------------------
__global__ void rf_kernel(const float* __restrict__ x,
                          const float* __restrict__ sc_w,
                          const float* __restrict__ sc_b,
                          const float* __restrict__ ac_w,
                          const float* __restrict__ ac_b,
                          _Float16* __restrict__ rfB) {
    __shared__ float tile[16][3][64];                 // 12 KB: 16 ch x 3 rows
    __shared__ float wsc[16 * 81];                    // 5184 B
    __shared__ float bsc[144], wac[144], bac[144];    // 1728 B
    __shared__ __align__(16) _Float16 stage[64 * SROW]; // 21504 B

    const int y = blockIdx.x, g = blockIdx.y, b = blockIdx.z;
    const int tid = threadIdx.x;

    for (int i = tid; i < 16 * 81; i += 256) wsc[i] = sc_w[g * 1296 + i];
    if (tid < 144) {
        bsc[tid] = sc_b[g * 144 + tid];
        wac[tid] = ac_w[g * 144 + tid];
        bac[tid] = ac_b[g * 144 + tid];
    }
    for (int i = tid; i < 16 * 3 * 64; i += 256) {
        int ci = i / 192, r = (i / 64) % 3, xc = i % 64;
        int gy = y - 1 + r;
        tile[ci][r][xc] = (gy >= 0 && gy < HH)
            ? x[((size_t)(b * CCH + g * 16 + ci)) * HW + gy * WW + xc] : 0.0f;
    }
    __syncthreads();

    const int xx = tid & 63;          // position within row
    const int cb = (tid >> 6) * 4;    // 4 channels per thread

    for (int cl = cb; cl < cb + 4; ++cl) {
        float w9[9];
        float ps = 0.0f;
        #pragma unroll
        for (int i = 0; i < 3; ++i)
            #pragma unroll
            for (int j = 0; j < 3; ++j) {
                int cx = xx + j - 1;
                float v = (cx >= 0 && cx < WW) ? tile[cl][i][cx] : 0.0f;
                w9[i * 3 + j] = v;
                ps += v;
            }
        float pooled = ps * (1.0f / 9.0f);

        float aa[9];
        float mx = -1e30f;
        #pragma unroll
        for (int q = 0; q < 9; ++q) {
            aa[q] = wac[cl * 9 + q] * pooled + bac[cl * 9 + q];
            mx = fmaxf(mx, aa[q]);
        }
        float se = 0.0f;
        #pragma unroll
        for (int q = 0; q < 9; ++q) { aa[q] = __expf(aa[q] - mx); se += aa[q]; }
        float inv = 1.0f / se;

        #pragma unroll
        for (int q = 0; q < 9; ++q) {
            float f = bsc[cl * 9 + q];
            #pragma unroll
            for (int t = 0; t < 9; ++t) f += wsc[cl * 81 + q * 9 + t] * w9[t];
            stage[xx * SROW + cl * 9 + q] = (_Float16)(silu_f(f) * aa[q] * inv);
        }
    }
    if (cb == 0) {
        #pragma unroll
        for (int k = 144; k < 160; ++k) stage[xx * SROW + k] = (_Float16)0.0f;
    }
    __syncthreads();

    // coalesced flush: 64 rows x 320 B, destination region is contiguous
    uint32_t* g32 = (uint32_t*)(rfB + ((size_t)(b * 4 + g) * HW + y * WW) * KPAD);
    const uint32_t* s32 = (const uint32_t*)stage;
    for (int i = tid; i < 64 * 20; i += 256) {
        int p = i / 20, ch = i % 20;
        *(uint4*)(g32 + p * 80 + ch * 4) = *(const uint4*)(s32 + p * 84 + ch * 4);
    }
}

// ---------------------------------------------------------------------------
// Kernel B1: per-(b,c) 2D FFT magnitude mean + spatial mean (LDS radix-2).
// ---------------------------------------------------------------------------
static __device__ void fft64(float* re, float* im, int stride) {
    for (int i = 0; i < 64; ++i) {
        int j = __brev((unsigned)i) >> 26;
        if (j > i) {
            float tr = re[i * stride]; re[i * stride] = re[j * stride]; re[j * stride] = tr;
            float ti = im[i * stride]; im[i * stride] = im[j * stride]; im[j * stride] = ti;
        }
    }
    for (int len = 2; len <= 64; len <<= 1) {
        int half = len >> 1;
        float ang0 = -6.28318530717958647692f / (float)len;
        for (int i0 = 0; i0 < 64; i0 += len) {
            for (int k = 0; k < half; ++k) {
                float sn, cs;
                __sincosf(ang0 * (float)k, &sn, &cs);
                int ia = (i0 + k) * stride, ib = (i0 + k + half) * stride;
                float br = re[ib], bi = im[ib];
                float vr = br * cs - bi * sn;
                float vi = br * sn + bi * cs;
                re[ib] = re[ia] - vr; im[ib] = im[ia] - vi;
                re[ia] += vr;         im[ia] += vi;
            }
        }
    }
}

__global__ void fft_energy_kernel(const float* __restrict__ x,
                                  float* __restrict__ meanx,
                                  float* __restrict__ energy) {
    __shared__ float re[HW];
    __shared__ float im[HW];
    __shared__ float red[64];

    const int c = blockIdx.x, b = blockIdx.y, t = threadIdx.x;
    const float* src = x + ((size_t)(b * CCH + c)) * HW;

    float s = 0.0f;
    for (int j = 0; j < 64; ++j) {
        float v = src[t * 64 + j];
        re[t * 64 + j] = v;
        im[t * 64 + j] = 0.0f;
        s += v;
    }
    red[t] = s;
    __syncthreads();
    for (int off = 32; off; off >>= 1) {
        if (t < off) red[t] += red[t + off];
        __syncthreads();
    }
    float msum = red[0];
    __syncthreads();

    fft64(&re[t * 64], &im[t * 64], 1);    // row t
    __syncthreads();
    fft64(&re[t], &im[t], 64);             // column t
    __syncthreads();

    float e = 0.0f;
    for (int j = 0; j < 64; ++j) {
        float rr = re[t * 64 + j], ii = im[t * 64 + j];
        e += sqrtf(rr * rr + ii * ii);
    }
    red[t] = e;
    __syncthreads();
    for (int off = 32; off; off >>= 1) {
        if (t < off) red[t] += red[t + off];
        __syncthreads();
    }
    if (t == 0) {
        meanx[b * CCH + c]  = msum * (1.0f / (float)HW);
        energy[b * CCH + c] = red[0] * (1.0f / (float)HW);
    }
}

// ---------------------------------------------------------------------------
// Kernel B2: per-batch channel-attention scalars -> coeff[b][c]
// ---------------------------------------------------------------------------
static __device__ __forceinline__ float conv3(const float* v, int c, const float* w) {
    float r = w[1] * v[c];
    if (c > 0)        r += w[0] * v[c - 1];
    if (c < CCH - 1)  r += w[2] * v[c + 1];
    return r;
}

__global__ void fgca_gate_kernel(const float* __restrict__ meanx,
                                 const float* __restrict__ energy,
                                 const float* __restrict__ ex_w,
                                 const float* __restrict__ ey_w,
                                 const float* __restrict__ ez_w,
                                 const float* __restrict__ ff_w,
                                 const float* __restrict__ ff_b,
                                 const float* __restrict__ g1_w,
                                 const float* __restrict__ g1_b,
                                 const float* __restrict__ g2_w,
                                 const float* __restrict__ g2_b,
                                 float* __restrict__ coeff) {
    __shared__ float mn[64], my[64], mz[64], en[64];
    __shared__ float cat[128], af[64], axs[64], gin[128], hbuf[16];

    const int b = blockIdx.x, t = threadIdx.x;
    mn[t] = meanx[b * CCH + t];
    en[t] = energy[b * CCH + t];
    __syncthreads();

    float et = en[t];
    int rank = 0;
    for (int d = 0; d < 64; ++d)
        rank += (en[d] > et) || (en[d] == et && d < t);
    float maskc = (rank < 32) ? 1.0f : 0.0f;
    my[t] = mn[t] * maskc;
    mz[t] = mn[t] * (1.0f - maskc);
    __syncthreads();

    float a_x = sigmoid_f(conv3(mn, t, ex_w));
    float a_y = sigmoid_f(conv3(my, t, ey_w));
    float a_z = sigmoid_f(conv3(mz, t, ez_w));
    cat[t] = a_y; cat[64 + t] = a_z; axs[t] = a_x;
    __syncthreads();

    {
        float v = ff_b[t];
        const int base = (t >> 4) * 32;
        for (int j = 0; j < 32; ++j) v += ff_w[t * 32 + j] * cat[base + j];
        af[t] = v; gin[t] = v; gin[64 + t] = a_x;
    }
    __syncthreads();

    if (t < 16) {
        float h = g1_b[t];
        for (int i = 0; i < 128; ++i) h += g1_w[t * 128 + i] * gin[i];
        hbuf[t] = silu_f(h);
    }
    __syncthreads();

    float gv = g2_b[t];
    for (int j = 0; j < 16; ++j) gv += g2_w[t * 16 + j] * hbuf[j];
    gv = sigmoid_f(gv);
    coeff[b * CCH + t] = gv * af[t] + (1.0f - gv) * axs[t];
}

// ---------------------------------------------------------------------------
// Kernel B3: x_fgca = x * coeff, LDS-transposed to coalesced 16B stores into
// fused channels 64..127. Block = (b, 64 consecutive positions).
// ---------------------------------------------------------------------------
__global__ void fgca_apply_kernel(const float* __restrict__ x,
                                  const float* __restrict__ coeff,
                                  _Float16* __restrict__ fused) {
    __shared__ float xin[64][65];                     // +1 pad: conflict-free transpose
    __shared__ __align__(16) _Float16 hst[64][64];    // [pos][ch] halves
    __shared__ float cf[64];

    const int b = blockIdx.y;
    const int p0 = blockIdx.x * 64;
    const int tid = threadIdx.x;

    if (tid < 64) cf[tid] = coeff[b * CCH + tid];
    for (int i = tid; i < 4096; i += 256) {
        int c = i >> 6, j = i & 63;
        xin[c][j] = x[((size_t)(b * CCH + c)) * HW + p0 + j];
    }
    __syncthreads();
    for (int i = tid; i < 4096; i += 256) {
        int p = i >> 6, c = i & 63;
        hst[p][c] = (_Float16)(xin[c][p] * cf[c]);
    }
    __syncthreads();
    char* base = (char*)fused + ((size_t)b * HW + p0) * (FCH * 2) + 128;
    for (int i = tid; i < 512; i += 256) {
        int p = i >> 3, ch = i & 7;
        *(uint4*)(base + (size_t)p * 256 + ch * 16) =
            *(const uint4*)((const char*)&hst[p][0] + ch * 16);
    }
}

// ---------------------------------------------------------------------------
// Kernel A2: grouped oc-conv as WMMA GEMM (16x160 x 160xN per group).
// A staged in LDS (async path when available); packed v8h epilogue stores.
// ---------------------------------------------------------------------------
__global__ void oc_gemm_kernel(const _Float16* __restrict__ rfB,
                               const _Float16* __restrict__ wA,
                               const float* __restrict__ oc_b,
                               const float* __restrict__ bn_g,
                               const float* __restrict__ bn_b,
                               const float* __restrict__ bn_m,
                               const float* __restrict__ bn_v,
                               _Float16* __restrict__ fused) {
    __shared__ __align__(16) _Float16 sA[16 * KPAD];  // 5120 B

    const int g    = blockIdx.y;
    const int b    = blockIdx.z;
    const int wave = threadIdx.x >> 5;
    const int lane = threadIdx.x & 31;
    const int pbase = (blockIdx.x * 8 + wave) * 16;

    stage_to_lds(wA + (size_t)g * 16 * KPAD, sA, 16 * KPAD, threadIdx.x, 256);

    const int row   = lane & 15;
    const int khalf = (lane >> 4) * 16;
    const _Float16* Bp = rfB + ((size_t)(b * 4 + g) * HW + pbase + row) * KPAD;

    v8f acc = {};
    #pragma unroll
    for (int kk = 0; kk < KPAD; kk += 32) {
        if (kk + 32 < KPAD) __builtin_prefetch(Bp + kk + 32 + khalf, 0, 1);
        v16h afrag = *(const v16h*)(&sA[row * KPAD + kk + khalf]);
        v16h bfrag = *(const v16h*)(Bp + kk + khalf);
        acc = __builtin_amdgcn_wmma_f32_16x16x32_f16(
                  false, afrag, false, bfrag, (short)0, acc, false, false);
    }

    // epilogue: M = (lane>>4)*8 + v, N = lane&15 -> one packed 16B store/lane
    const int p = pbase + (lane & 15);
    const int ob = g * 16 + (lane >> 4) * 8;
    v8h o8;
    #pragma unroll
    for (int v = 0; v < 8; ++v) {
        int och = ob + v;
        float s   = bn_g[och] * rsqrtf(bn_v[och] + 1e-5f);
        float val = (acc[v] + oc_b[och]) * s + (bn_b[och] - bn_m[och] * s);
        o8[v] = (_Float16)silu_f(val);
    }
    *(v8h*)(&fused[((size_t)b * HW + p) * FCH + ob]) = o8;
}

// ---------------------------------------------------------------------------
// Kernel C: fuse 1x1 conv as WMMA GEMM (32x128 x 128xN).
// B fragments hoisted across both M tiles; LDS-transposed f32 epilogue.
// ---------------------------------------------------------------------------
__global__ void fu_gemm_kernel(const _Float16* __restrict__ fused,
                               const _Float16* __restrict__ wF,
                               const float* __restrict__ fu_b,
                               const float* __restrict__ bn_g,
                               const float* __restrict__ bn_b,
                               const float* __restrict__ bn_m,
                               const float* __restrict__ bn_v,
                               float* __restrict__ out) {
    __shared__ __align__(16) _Float16 sF[OUTC * FCH]; // 8192 B
    __shared__ float dtile[8][16][20];                // 10240 B, wave-private tiles

    const int b    = blockIdx.y;
    const int wave = threadIdx.x >> 5;
    const int lane = threadIdx.x & 31;
    const int pbase = (blockIdx.x * 8 + wave) * 16;

    stage_to_lds(wF, sF, OUTC * FCH, threadIdx.x, 256);

    const int row   = lane & 15;
    const int khalf = (lane >> 4) * 16;

    const _Float16* Bp = fused + ((size_t)b * HW + pbase + row) * FCH;
    v16h bfr[4];
    #pragma unroll
    for (int i = 0; i < 4; ++i)
        bfr[i] = *(const v16h*)(Bp + i * 32 + khalf);

    #pragma unroll
    for (int mt = 0; mt < 2; ++mt) {
        v8f acc = {};
        #pragma unroll
        for (int i = 0; i < 4; ++i) {
            v16h afrag = *(const v16h*)(&sF[(mt * 16 + row) * FCH + i * 32 + khalf]);
            acc = __builtin_amdgcn_wmma_f32_16x16x32_f16(
                      false, afrag, false, bfr[i], (short)0, acc, false, false);
        }
        // epilogue with in-wave LDS transpose -> 32B contiguous stores per lane
        #pragma unroll
        for (int v = 0; v < 8; ++v) {
            int M = (lane >> 4) * 8 + v;
            int o = mt * 16 + M;
            float s   = bn_g[o] * rsqrtf(bn_v[o] + 1e-5f);
            float val = (acc[v] + fu_b[o]) * s + (bn_b[o] - bn_m[o] * s);
            dtile[wave][M][lane & 15] = silu_f(val);
        }
        int o_loc = lane >> 1;
        int xoff  = (lane & 1) * 8;
        float4 r0 = *(const float4*)&dtile[wave][o_loc][xoff];
        float4 r1 = *(const float4*)&dtile[wave][o_loc][xoff + 4];
        float* op = out + ((size_t)b * OUTC + mt * 16 + o_loc) * HW + pbase + xoff;
        *(float4*)op = r0;
        *(float4*)(op + 4) = r1;
    }
}

// ---------------------------------------------------------------------------
// Launch
// ---------------------------------------------------------------------------
extern "C" void kernel_launch(void* const* d_in, const int* in_sizes, int n_in,
                              void* d_out, int out_size, void* d_ws, size_t ws_size,
                              hipStream_t stream) {
    const float* x     = (const float*)d_in[0];
    const float* sc_w  = (const float*)d_in[1];
    const float* sc_b  = (const float*)d_in[2];
    const float* ac_w  = (const float*)d_in[3];
    const float* ac_b  = (const float*)d_in[4];
    const float* oc_w  = (const float*)d_in[5];
    const float* oc_b  = (const float*)d_in[6];
    const float* ocg   = (const float*)d_in[7];
    const float* ocb   = (const float*)d_in[8];
    const float* ocm   = (const float*)d_in[9];
    const float* ocv   = (const float*)d_in[10];
    const float* ex_w  = (const float*)d_in[11];
    const float* ey_w  = (const float*)d_in[12];
    const float* ez_w  = (const float*)d_in[13];
    const float* ff_w  = (const float*)d_in[14];
    const float* ff_b  = (const float*)d_in[15];
    const float* g1_w  = (const float*)d_in[16];
    const float* g1_b  = (const float*)d_in[17];
    const float* g2_w  = (const float*)d_in[18];
    const float* g2_b  = (const float*)d_in[19];
    const float* fu_w  = (const float*)d_in[20];
    const float* fu_b  = (const float*)d_in[21];
    const float* fug   = (const float*)d_in[22];
    const float* fub   = (const float*)d_in[23];
    const float* fum   = (const float*)d_in[24];
    const float* fuv   = (const float*)d_in[25];

    // workspace layout (all 32B aligned)
    char* ws = (char*)d_ws;
    _Float16* rfB   = (_Float16*)(ws);                 // 8*4*4096*160 f16 = 41,943,040 B
    _Float16* fused = (_Float16*)(ws + 41943040ULL);   //   8*4096*128 f16 =  8,388,608 B
    _Float16* wA    = (_Float16*)(ws + 50331648ULL);   //      4*16*160 f16 =    20,480 B
    _Float16* wF    = (_Float16*)(ws + 50352128ULL);   //        32*128 f16 =     8,192 B
    float*    meanx = (float*)   (ws + 50360320ULL);   //           512 f32
    float*    energy = meanx + 512;
    float*    coeff  = energy + 512;

    prep_weights_kernel<<<1, 256, 0, stream>>>(oc_w, fu_w, wA, wF);

    rf_kernel<<<dim3(HH, 4, BSZ), 256, 0, stream>>>(
        x, sc_w, sc_b, ac_w, ac_b, rfB);

    fft_energy_kernel<<<dim3(CCH, BSZ), 64, 0, stream>>>(x, meanx, energy);

    fgca_gate_kernel<<<BSZ, 64, 0, stream>>>(
        meanx, energy, ex_w, ey_w, ez_w, ff_w, ff_b,
        g1_w, g1_b, g2_w, g2_b, coeff);

    fgca_apply_kernel<<<dim3(HW / 64, BSZ), 256, 0, stream>>>(x, coeff, fused);

    oc_gemm_kernel<<<dim3(HW / 128, 4, BSZ), 256, 0, stream>>>(
        rfB, wA, oc_b, ocg, ocb, ocm, ocv, fused);

    fu_gemm_kernel<<<dim3(HW / 128, BSZ), 256, 0, stream>>>(
        fused, wF, fu_b, fug, fub, fum, fuv, (float*)d_out);
}
